// HierarchicalBiologicalSplatAttentionLayer_49744311222306
// MI455X (gfx1250) — matrix operand order repaired and checked by reading, not compiled, not run on hardware
//
#include <hip/hip_runtime.h>
#include <hip/hip_bf16.h>

typedef __attribute__((ext_vector_type(16))) __bf16 v16bf;
typedef __attribute__((ext_vector_type(8)))  float  v8f;
typedef __attribute__((ext_vector_type(2)))  float  v2f;
typedef __attribute__((ext_vector_type(4)))  int    v4i;

#define BATCH 4
#define TOK   8192
#define DIM   512
#define NS    16
#define BT    (BATCH*TOK)
#define PAD   516   // LDS row stride in floats (+4 to break bank conflicts)

// ---- CDNA5 async global->LDS staging (ASYNCcnt path), with synchronous fallback ----
#if defined(__gfx1250__) && __has_builtin(__builtin_amdgcn_global_load_async_to_lds_b128) && __has_builtin(__builtin_amdgcn_s_wait_asynccnt)
#define HAS_ASYNC_LDS 1
__device__ __forceinline__ void async_ld_b128(const float* g, float* l) {
    // builtin expects pointers to 128-bit int vectors (per hipcc diagnostic)
    __builtin_amdgcn_global_load_async_to_lds_b128(
        (v4i*)(size_t)g,   // int->ptr cast strips const, keeps flat address
        (v4i*)(void*)l,
        0, 0);
}
__device__ __forceinline__ void async_wait0() { __builtin_amdgcn_s_wait_asynccnt(0); }
#else
#define HAS_ASYNC_LDS 0
#endif

// ---------------- prep: Wv -> bf16, zero ss accumulator, c2 & 1/sigma^2 ----------------
__global__ __launch_bounds__(256) void k_prep(const float* __restrict__ Wv,
                                              const float* __restrict__ centers,
                                              const float* __restrict__ log_scales,
                                              __bf16* __restrict__ Wv_b,
                                              float*  __restrict__ c2inv,
                                              float*  __restrict__ ss)
{
    const int tid    = blockIdx.x * blockDim.x + threadIdx.x;
    const int stride = gridDim.x * blockDim.x;
    for (int i = tid; i < DIM * DIM; i += stride) Wv_b[i] = (__bf16)Wv[i];
    for (int i = tid; i < BATCH * NS * DIM; i += stride) ss[i] = 0.0f;  // re-zero every call
    if (tid < NS) {
        float c2 = 0.0f;
        const float* c = centers + tid * DIM;
        for (int d = 0; d < DIM; ++d) c2 = fmaf(c[d], c[d], c2);
        float sc = __expf(log_scales[tid]);
        sc = fminf(fmaxf(sc, 0.1f), 2.0f);
        c2inv[tid]      = c2;                 // ||c_s||^2
        c2inv[NS + tid] = 1.0f / (sc * sc);   // 1/sigma_s^2
    }
}

// ---------------- K1: fused  v = x@Wv^T (bf16 WMMA), xc (f32 WMMA, all 8 waves),
//                  softmax, partial splat_states += attn^T @ v (bf16 WMMA + atomics) ----
__global__ __launch_bounds__(256) void k_gemm1(const float* __restrict__ x,
                                               const float* __restrict__ centers,
                                               const __bf16* __restrict__ Wv_b,
                                               const float* __restrict__ c2inv,
                                               float* __restrict__ attn_g,
                                               float* __restrict__ ss)
{
    __shared__ float xv_s[16 * PAD];   // x tile, later reused as v tile (fp32)
    __shared__ float xcp_s[8 * 256];   // per-wave partial xc tiles [wave][token][splat]
    __shared__ float at_s[16 * 16];    // attn tile [token][splat]
    __shared__ float x2p_s[16 * 16];   // ||x||^2 partials [token][chunk]

    const int tid  = threadIdx.x;
    const int lane = tid & 31;
    const int wave = tid >> 5;         // 8 waves
    const int hi   = lane >> 4;        // lane half (0/1)
    const int lo   = lane & 15;
    const int g0   = blockIdx.x * 16;  // first global token row of this tile
    const int b    = g0 >> 13;         // g0 / 8192
    const float* xt = x + (size_t)g0 * DIM;

    // stage 16x512 x tile into LDS
#if HAS_ASYNC_LDS
    for (int i = tid; i < 16 * (DIM / 4); i += 256) {
        const int t = i >> 7;
        const int c = (i & 127) << 2;
        async_ld_b128(xt + (size_t)t * DIM + c, &xv_s[t * PAD + c]);
    }
    async_wait0();
#else
    for (int i = tid; i < 16 * (DIM / 4); i += 256) {
        const int t = i >> 7;
        const int c = (i & 127) << 2;
        float4 v = *(const float4*)(xt + (size_t)t * DIM + c);
        *(float4*)(&xv_s[t * PAD + c]) = v;
    }
#endif
    __syncthreads();

    // ||x_t||^2 partial sums: thread = (token, 32-wide chunk)
    {
        const int t = tid >> 4, c = tid & 15;
        const float* p = &xv_s[t * PAD + c * 32];
        float s = 0.0f;
        for (int i = 0; i < 32; ++i) s = fmaf(p[i], p[i], s);
        x2p_s[tid] = s;  // == x2p_s[t*16 + c]
    }

    // ---- v = x @ Wv^T : each wave owns 64 output columns = 4 WMMA N-tiles ----
    v8f acc[4] = {};
    const int n0 = wave * 64;
    for (int k0 = 0; k0 < DIM; k0 += 32) {
        // A (16x32 bf16): lane lo = token row; elem i -> K=hi*16+i
        v16bf a;
        const float* pa = &xv_s[lo * PAD + k0 + hi * 16];
        #pragma unroll
        for (int i = 0; i < 8; ++i) a[i] = (__bf16)pa[i];
        #pragma unroll
        for (int i = 0; i < 8; ++i) a[8 + i] = (__bf16)pa[8 + i];
        // B (32x16 bf16): B[k][n] = Wv[e=n][d=k]; lane col e, 16 contiguous d per lane
        const int dB = k0 + hi * 16;
        #pragma unroll
        for (int j = 0; j < 4; ++j) {
            const v16bf bm = *(const v16bf*)(Wv_b + (size_t)(n0 + j * 16 + lo) * DIM + dB);
            acc[j] = __builtin_amdgcn_wmma_f32_16x16x32_bf16(false, a, false, bm,
                                                             (short)0, acc[j], false, false);
        }
    }

    // ---- xc = x @ centers^T in full fp32: K split across 8 waves (16 steps each) ----
    {
        v8f xca = {};
        const int kbase = wave * 64;
        for (int k0 = kbase; k0 < kbase + 64; k0 += 4) {
            v2f av, bv;
            const float* pa = &xv_s[lo * PAD + k0 + hi * 2];            // A: token row lo
            av[0] = pa[0]; av[1] = pa[1];
            const float* pc = centers + (size_t)lo * DIM + k0 + hi * 2; // B: splat col lo
            bv[0] = pc[0]; bv[1] = pc[1];
            xca = __builtin_amdgcn_wmma_f32_16x16x4_f32(false, av, false, bv,
                                                        (short)0, xca, false, false);
        }
        #pragma unroll
        for (int r = 0; r < 8; ++r)
            xcp_s[wave * 256 + (r + hi * 8) * 16 + lo] = xca[r];
    }
    __syncthreads();  // x staging dead from here -> reuse xv_s for v

    // spill v tile (fp32) into LDS for the cross-lane transpose needed by the ss WMMA
    #pragma unroll
    for (int j = 0; j < 4; ++j) {
        #pragma unroll
        for (int r = 0; r < 8; ++r)
            xv_s[(r + hi * 8) * PAD + n0 + j * 16 + lo] = acc[j][r];
    }

    // gaussian logits + softmax over the 16 splats (one thread per token, deterministic)
    if (tid < 16) {
        float x2 = 0.0f;
        #pragma unroll
        for (int c = 0; c < 16; ++c) x2 += x2p_s[tid * 16 + c];
        float lg[NS]; float mx = -3.0e38f;
        #pragma unroll
        for (int s = 0; s < NS; ++s) {
            float xc = 0.0f;
            #pragma unroll
            for (int w = 0; w < 8; ++w) xc += xcp_s[w * 256 + tid * 16 + s];
            // -0.5*(x2 + c2 - 2*xc)/sigma^2
            float l = (xc - 0.5f * (x2 + c2inv[s])) * c2inv[NS + s];
            lg[s] = l; mx = fmaxf(mx, l);
        }
        float sum = 0.0f;
        #pragma unroll
        for (int s = 0; s < NS; ++s) { float e = __expf(lg[s] - mx); lg[s] = e; sum += e; }
        const float inv = 1.0f / sum;
        #pragma unroll
        for (int s = 0; s < NS; ++s) {
            const float aw = lg[s] * inv;
            at_s[tid * 16 + s] = aw;
            attn_g[((size_t)g0 + tid) * NS + s] = aw;
        }
    }
    __syncthreads();

    // ---- partial splat_states: ssp = attn^T (16s x 16t, zero-padded K=32) @ v ----
    v16bf aA;
    {
        #pragma unroll
        for (int i = 0; i < 8; ++i) aA[i] = hi ? (__bf16)0.0f : (__bf16)at_s[i * 16 + lo]; // K=token 0..7 (hi=0)
        #pragma unroll
        for (int i = 0; i < 8; ++i) aA[8 + i] = hi ? (__bf16)0.0f : (__bf16)at_s[(8 + i) * 16 + lo]; // K=token 8..15 (hi=0); hi=1: pad K 16..31
    }
    float* ssb = ss + (size_t)b * NS * DIM;
    #pragma unroll
    for (int j = 0; j < 4; ++j) {
        const int d = n0 + j * 16 + lo;
        v16bf bB;
        if (hi == 0) {   // lanes 0..15: K=0..15 = real tokens
            #pragma unroll
            for (int i = 0; i < 16; ++i) bB[i] = (__bf16)xv_s[i * PAD + d];
        } else {         // lanes 16..31: K=16..31 = zero pad
            #pragma unroll
            for (int i = 0; i < 16; ++i) bB[i] = (__bf16)0.0f;
        }
        v8f sa = {};
        sa = __builtin_amdgcn_wmma_f32_16x16x32_bf16(false, aA, false, bB,
                                                     (short)0, sa, false, false);
        #pragma unroll
        for (int r = 0; r < 8; ++r)
            atomicAdd(&ssb[(r + hi * 8) * DIM + d], sa[r]);  // global_atomic_add_f32
    }
}

// ---------------- K2: M[b] = ss[b] @ Wo^T   (tiny: 4x 16x512x512 fp32) ----------------
__global__ __launch_bounds__(256) void k_combine(const float* __restrict__ ss,
                                                 const float* __restrict__ Wo,
                                                 float* __restrict__ M)
{
    __shared__ float ss_s[NS * DIM];
    const int b  = blockIdx.x >> 4;
    const int e0 = (blockIdx.x & 15) * 32;
    const int tid = threadIdx.x;
    for (int i = tid; i < NS * DIM; i += 256) ss_s[i] = ss[(size_t)b * NS * DIM + i];
    __syncthreads();
    for (int p = tid; p < NS * 32; p += 256) {
        const int s = p >> 5;
        const int e = e0 + (p & 31);
        const float* w = Wo + (size_t)e * DIM;
        const float* v = ss_s + s * DIM;
        float acc = 0.0f;
        for (int d = 0; d < DIM; ++d) acc = fmaf(v[d], w[d], acc);
        M[((size_t)b * NS + s) * DIM + e] = acc;
    }
}

// ---------------- K3: final[t,:] = attn[t,:] @ M[b]  (fp32, coalesced 64MB write) ------
__global__ __launch_bounds__(256) void k_final(const float* __restrict__ attn,
                                               const float* __restrict__ M,
                                               float* __restrict__ out)
{
    __shared__ float M_s[NS * DIM];
    __shared__ float a_s[16 * NS];
    const int tid = threadIdx.x;
    const int g0  = blockIdx.x * 16;
    const int b   = g0 >> 13;
    const float* Mb = M + (size_t)b * NS * DIM;
#if HAS_ASYNC_LDS
    for (int i = tid; i < NS * DIM / 4; i += 256)
        async_ld_b128(Mb + i * 4, &M_s[i * 4]);
#else
    for (int i = tid; i < NS * DIM; i += 256) M_s[i] = Mb[i];
#endif
    for (int i = tid; i < 16 * NS; i += 256)  a_s[i] = attn[(size_t)g0 * NS + i];
#if HAS_ASYNC_LDS
    async_wait0();
#endif
    __syncthreads();
    for (int p = tid; p < 16 * DIM; p += 256) {
        const int t = p >> 9;
        const int e = p & 511;
        float acc = 0.0f;
        #pragma unroll
        for (int s = 0; s < NS; ++s) acc = fmaf(a_s[t * NS + s], M_s[s * DIM + e], acc);
        out[((size_t)g0 + t) * DIM + e] = acc;
    }
}

extern "C" void kernel_launch(void* const* d_in, const int* in_sizes, int n_in,
                              void* d_out, int out_size, void* d_ws, size_t ws_size,
                              hipStream_t stream)
{
    const float* x       = (const float*)d_in[0];  // [B,T,D]
    const float* centers = (const float*)d_in[1];  // [S,D]
    const float* logsc   = (const float*)d_in[2];  // [S]
    const float* Wv      = (const float*)d_in[3];  // [D,D]
    const float* Wo      = (const float*)d_in[4];  // [D,D]
    float* out = (float*)d_out;

    // workspace carve (all 256B-aligned): Wv_bf16 | c2+inv_s2 | ss | M | attn
    char* w = (char*)d_ws;
    __bf16* Wv_b = (__bf16*)(w);                                   // 512 KB
    float*  c2i  = (float*) (w + 524288);                          // 256 B slot
    float*  ss   = (float*) (w + 524288 + 256);                    // 128 KB
    float*  M    = (float*) (w + 524288 + 256 + 131072);           // 128 KB
    float*  attn = (float*) (w + 524288 + 256 + 131072 + 131072);  // 2 MB

    k_prep   <<<256,        256, 0, stream>>>(Wv, centers, logsc, Wv_b, c2i, ss);
    k_gemm1  <<<BT / 16,    256, 0, stream>>>(x, centers, Wv_b, c2i, attn, ss);
    k_combine<<<BATCH * 16, 256, 0, stream>>>(ss, Wo, M);
    k_final  <<<BT / 16,    256, 0, stream>>>(attn, M, out);
}
